// PredecessorDecoder_87299505258630
// MI455X (gfx1250) — matrix operand order repaired and checked by reading, not compiled, not run on hardware
//
#include <hip/hip_runtime.h>

typedef __attribute__((ext_vector_type(16))) _Float16     v16h;
typedef __attribute__((ext_vector_type(2)))  _Float16     h2v;
typedef __attribute__((ext_vector_type(8)))  float        v8f;
typedef __attribute__((ext_vector_type(4)))  unsigned int u32x4;
typedef __attribute__((ext_vector_type(8)))  unsigned int u32x8;

#define N_NODES 1024
#define LATENT  64
#define NFEAT   128          // concat(encoded, h) per node
#define ROWP    66           // padded LDS row stride in halves (132 B; gcd(33,64)=1 -> conflict-free)
#define S2_BLOCK 256

// ---------------------------------------------------------------------------
// Stage 1: per-node precompute via WMMA (f16 inputs, f32 accumulate).
//   P[0][n][:] = [enc[n]|h[n]] @ W1[0:128,:]   + b1   (src contribution)
//   P[1][n][:] = [enc[n]|h[n]] @ W1[128:256,:]        (dst contribution)
// One 16x16 output tile per wave; K=128 -> 4 x v_wmma_f32_16x16x32_f16.
// Fragment lane layouts per CDNA5 ISA 7.12.2.
// ---------------------------------------------------------------------------
__global__ __launch_bounds__(128)
void node_precompute_wmma(const float* __restrict__ encoded,
                          const float* __restrict__ h,
                          const float* __restrict__ W1,
                          const float* __restrict__ b1,
                          _Float16* __restrict__ P)
{
    const int lane      = threadIdx.x & 31;
    const int waveInBlk = threadIdx.x >> 5;
    const int tile      = blockIdx.x * 4 + waveInBlk;   // 0..511
    const int t         = tile >> 8;                    // table 0 (src) / 1 (dst)
    const int rem       = tile & 255;
    const int mt        = rem >> 2;                     // node tile 0..63
    const int nt        = rem & 3;                      // output-col tile 0..3

    const int half16 = lane >> 4;                       // which 16-lane group
    const int l16    = lane & 15;
    const int m      = mt * 16 + l16;                   // A-frag row (node)
    const int n      = nt * 16 + l16;                   // B-frag col / C col

    v8f acc = {};
    #pragma unroll
    for (int kc = 0; kc < 4; ++kc) {
        const int kbase = kc * 32;
        v16h a, b;
        #pragma unroll
        for (int j = 0; j < 16; ++j) {
            // A 16x32 f16 layout: VGPR v<4 holds K=2v,2v+1 (+8 for lanes>=16),
            //                     VGPR v>=4 holds K=8+2v.. (+8 for lanes>=16)
            const int v     = j >> 1;
            const int klocA = (v < 4 ? 2 * v : 8 + 2 * v) + 8 * half16 + (j & 1);
            const int ka    = kbase + klocA;
            const float av  = (ka < LATENT) ? encoded[m * LATENT + ka]
                                            : h[m * LATENT + (ka - LATENT)];
            a[j] = (_Float16)av;
            // B 32x16 f16 layout: same K mapping as A (K=0..7,16..23 lanes 0-15; K=8..15,24..31 lanes 16-31)
            const int kb = kbase + klocA;
            b[j] = (_Float16)W1[(t * NFEAT + kb) * LATENT + n];
        }
        acc = __builtin_amdgcn_wmma_f32_16x16x32_f16(
                  /*neg_a=*/false, a, /*neg_b=*/false, b,
                  /*c_mod=*/(short)0, acc, /*reuse_a=*/false, /*reuse_b=*/false);
    }

    const float bias = (t == 0) ? b1[n] : 0.0f;
    // C/D layout: VGPR r -> M = r + 8*half16, N = l16
    #pragma unroll
    for (int r = 0; r < 8; ++r) {
        const int mOut = mt * 16 + r + 8 * half16;
        P[(size_t)t * (N_NODES * LATENT) + mOut * LATENT + n] =
            (_Float16)(acc[r] + bias);
    }
}

// ---------------------------------------------------------------------------
// TDM: DMA one 1024x64 f16 table from global memory into LDS, inserting one
// DWORD of padding after every row (32 DWORDs) -> 132-B padded rows in LDS.
// D# layout per CDNA5 ISA 8.3 (group 0) / 8.4 (group 1).
// ---------------------------------------------------------------------------
__device__ __forceinline__ void tdm_load_table(const _Float16* gsrc, void* lds_dst)
{
    const unsigned long long ga = (unsigned long long)(uintptr_t)gsrc;
    u32x4 g0;
    g0[0] = 1u;                                        // count=1, user D#
    g0[1] = (unsigned)(uintptr_t)lds_dst;              // lds_addr (bytes)
    g0[2] = (unsigned)ga;                              // global_addr[31:0]
    g0[3] = ((unsigned)(ga >> 32) & 0x01FFFFFFu)       // global_addr[56:32]
          | 0x80000000u;                               // type=2 ("image")
    u32x8 g1;
    g1[0] = (1u << 16)                                 // data_size = 2 bytes
          | (1u << 20)                                 // pad_enable
          | (4u << 22);                                // pad_interval: 32 DWORDs; pad_amount=0 -> 1 DWORD
    g1[1] = (LATENT & 0xFFFFu) << 16;                  // tensor_dim0 = 64  (bits 79:48)
    g1[2] = (LATENT >> 16) | ((N_NODES & 0xFFFFu) << 16); // tensor_dim1 = 1024 (bits 111:80)
    g1[3] = (N_NODES >> 16) | ((unsigned)LATENT << 16);   // tile_dim0 = 64 (bits 127:112)
    g1[4] = N_NODES;                                   // tile_dim1 = 1024; tile_dim2 = 0
    g1[5] = LATENT;                                    // tensor_dim0_stride = 64
    g1[6] = 0u;                                        // stride0 hi / tensor_dim1_stride lo
    g1[7] = 0u;
    asm volatile("tensor_load_to_lds %0, %1" :: "s"(g0), "s"(g1) : "memory");
}

// ---------------------------------------------------------------------------
// Stage 2: TDM-stage both 1024x64 f16 tables into LDS (padded rows), then per
// edge: out[e] = relu(A[src]+B[dst]) . W2 + b2. Gathers served from LDS
// instead of ~512 MB of random L2 reads.
// ---------------------------------------------------------------------------
__global__ __launch_bounds__(S2_BLOCK)
void edge_decode(const int* __restrict__ edge_index,   // [2*E]
                 const float* __restrict__ W2,
                 const float* __restrict__ b2,
                 const _Float16* __restrict__ P,
                 float* __restrict__ out,
                 int E, int chunk)
{
    extern __shared__ char smem[];
    _Float16* sA = (_Float16*)smem;                    // N_NODES * ROWP halves
    _Float16* sB = sA + N_NODES * ROWP;
    float*    sW = (float*)(sB + N_NODES * ROWP);      // 64 floats

    // Wave 0 (wave-uniform branch, full EXEC) issues both table DMAs.
    if (threadIdx.x < 32) {
        tdm_load_table(P, sA);
        tdm_load_table(P + N_NODES * LATENT, sB);
        __builtin_amdgcn_s_wait_tensorcnt(0);
    }
    if (threadIdx.x < LATENT) sW[threadIdx.x] = W2[threadIdx.x];
    __syncthreads();

    const float bias2 = b2[0];
    const int base = blockIdx.x * chunk;
    for (int i = threadIdx.x; i < chunk; i += S2_BLOCK) {
        const int e = base + i;
        if (e >= E) break;
        __builtin_prefetch(&edge_index[e + S2_BLOCK], 0, 0);       // index stream
        __builtin_prefetch(&edge_index[E + e + S2_BLOCK], 0, 0);
        const int s = edge_index[e];
        const int d = edge_index[E + e];
        const _Float16* ra = &sA[s * ROWP];
        const _Float16* rb = &sB[d * ROWP];
        float acc = bias2;
        #pragma unroll
        for (int k = 0; k < LATENT; k += 2) {
            const h2v va = *(const h2v*)(&ra[k]);      // 4B-aligned ds_load
            const h2v vb = *(const h2v*)(&rb[k]);
            float x0 = (float)va.x + (float)vb.x;
            float x1 = (float)va.y + (float)vb.y;
            x0 = x0 > 0.0f ? x0 : 0.0f;
            x1 = x1 > 0.0f ? x1 : 0.0f;
            acc = fmaf(x0, sW[k],     acc);
            acc = fmaf(x1, sW[k + 1], acc);
        }
        out[e] = acc;
    }
}

// ---------------------------------------------------------------------------
extern "C" void kernel_launch(void* const* d_in, const int* in_sizes, int n_in,
                              void* d_out, int out_size, void* d_ws, size_t ws_size,
                              hipStream_t stream) {
    const float* encoded = (const float*)d_in[0];
    const float* hfeat   = (const float*)d_in[1];
    const int*   eidx    = (const int*)d_in[2];
    const float* W1      = (const float*)d_in[3];
    const float* b1      = (const float*)d_in[4];
    const float* W2      = (const float*)d_in[5];
    const float* b2      = (const float*)d_in[6];

    const int E = in_sizes[2] / 2;

    _Float16* P = (_Float16*)d_ws;   // 2 * 1024 * 64 halves = 256 KB scratch

    // Stage 1: 512 tiles (2 tables x 64 m-tiles x 4 n-tiles), 1 tile/wave.
    node_precompute_wmma<<<128, 128, 0, stream>>>(encoded, hfeat, W1, b1, P);

    // Stage 2: 256 blocks, each TDM-stages full tables to LDS then decodes a chunk.
    const int grid  = 256;
    const int chunk = (E + grid - 1) / grid;
    const size_t shmem = (size_t)(2 * N_NODES * ROWP) * sizeof(_Float16)
                       + LATENT * sizeof(float);
    edge_decode<<<grid, S2_BLOCK, shmem, stream>>>(eidx, W2, b2, P,
                                                   (float*)d_out, E, chunk);
}
